// STMGFN_22617297781209
// MI455X (gfx1250) — compile-verified, matrix operand order checked by reading, not run on hardware
//
#include <hip/hip_runtime.h>
#include <math.h>

// ---------------------------------------------------------------------------
// MI455X / gfx1250, wave32. Diffusion-GCN GRU cell.
//   dominant work: 16x [4096x4096] x [4096x1280] GEMMs -> v_wmma bf16 path
//   A staged f32->bf16 via float4 loads + packed cvt; double-buffered LDS,
//   software-pipelined global loads; B fragments preloaded AND pinned via
//   empty asm ties so the WMMAs issue as one burst behind a single
//   s_wait_dscnt; coalesced b64 epilogue via LDS.
// ---------------------------------------------------------------------------

typedef __attribute__((ext_vector_type(16))) __bf16 v16bf;
typedef __attribute__((ext_vector_type(8)))  __bf16 v8bf;
typedef __attribute__((ext_vector_type(4)))  __bf16 v4bf;
typedef __attribute__((ext_vector_type(8)))  float  v8f;

constexpr int B_   = 16;
constexpr int N_   = 4096;
constexpr int DIN  = 16;
constexpr int DOUT = 64;
constexpr int S_   = 4;
constexpr int CAT  = DIN + DOUT;         // 80
constexpr int CIN  = (2 * S_ + 1) * CAT; // 720
constexpr int BN   = B_ * N_;            // 65536
constexpr int LDK  = 48;                 // LDS row stride (bf16); 96B keeps 16B frag alignment

// Assemble a 16x16x32 bf16 WMMA fragment from LDS.
// ISA layout (16-bit A/B 16x32): lanes 0-15 hold K {0..7, 16..23}, lanes 16-31
// hold K {8..15, 24..31} -> two contiguous 16B chunks per lane.
__device__ __forceinline__ v16bf frag_ld(const __bf16* rowbase, int koff) {
    v8bf lo = *(const v8bf*)(rowbase + koff);
    v8bf hi = *(const v8bf*)(rowbase + 16 + koff);
    v16bf r;
#pragma unroll
    for (int i = 0; i < 8; ++i) { r[i] = lo[i]; r[8 + i] = hi[i]; }
    return r;
}

// Scheduling pin: forces the fragment to be materialized HERE, so the backend
// cannot sink its ds_loads down between the WMMAs (which would re-introduce a
// s_wait_dscnt per WMMA instead of one wait per K-step).
__device__ __forceinline__ void pin(v16bf& v) {
    asm volatile("" : "+v"(v));
}

__device__ __forceinline__ v4bf cvt4(float4 f) {
    v4bf p;
    p[0] = (__bf16)f.x; p[1] = (__bf16)f.y;
    p[2] = (__bf16)f.z; p[3] = (__bf16)f.w;
    return p;
}

__device__ __forceinline__ v4bf zero4() {
    v4bf z; z[0] = z[1] = z[2] = z[3] = (__bf16)0.0f; return z;
}

// ---------------------------------------------------------------------------
// pack: concat chunk0 of H = [x || state] (or [x || z*state]) as bf16
// ---------------------------------------------------------------------------
template<bool WITHZ>
__global__ __launch_bounds__(256) void pack_kernel(
    const float* __restrict__ x, const float* __restrict__ state,
    const float* __restrict__ ZR, __bf16* __restrict__ H) {
    int idx = blockIdx.x * 256 + threadIdx.x;
    if (idx >= BN * CAT) return;
    int row = idx / CAT;
    int c   = idx - row * CAT;
    float v;
    if (c < DIN) {
        v = x[(size_t)row * DIN + c];
    } else {
        float s = state[(size_t)row * DOUT + (c - DIN)];
        if (WITHZ) v = ZR[(size_t)row * 128 + (c - DIN)] * s;   // z = ZR[:, :64]
        else       v = s;
    }
    H[(size_t)row * CIN + c] = (__bf16)v;
}

// ---------------------------------------------------------------------------
// prop_gemm: for batch b,  Hdst[b, :, dst_c:+80] = A @ Hsrc[b, :, src_c:+80]
//   block tile 128 x 80, 8 waves; wave = 16 rows x 5 wmma tiles
// ---------------------------------------------------------------------------
__global__ __launch_bounds__(256) void prop_gemm(
    const float* __restrict__ A, const __bf16* __restrict__ Hsrc,
    __bf16* __restrict__ Hdst, int src_c, int dst_c) {
    __shared__ __align__(16) __bf16 smem[2 * 128 * LDK + 2 * CAT * LDK]; // 39.9KB
    __bf16* AsBuf = smem;                    // [2][128*LDK]
    __bf16* BtBuf = smem + 2 * 128 * LDK;    // [2][80*LDK]

    const int tid  = threadIdx.x;
    const int wv   = tid >> 5;
    const int ln   = tid & 31;
    const int hl   = ln >> 4;
    const int koff = hl * 8;
    const int m0   = blockIdx.x * 128;
    const size_t bofs = (size_t)blockIdx.y * N_;

    float4 ra[4];   // staged A: 128x32 f32 = 1024 float4 slots, 4/thread
    v4bf   rb[3];   // staged B: 32x80 bf16 = 640 v4bf slots, <=3/thread

    auto gload = [&](int k0) {
#pragma unroll
        for (int s = 0; s < 4; ++s) {
            int slot = tid + s * 256;               // 0..1023
            int r = slot >> 3, k4 = slot & 7;       // 8 float4 per 32-wide row
            ra[s] = *(const float4*)&A[(size_t)(m0 + r) * N_ + k0 + k4 * 4];
        }
#pragma unroll
        for (int s = 0; s < 3; ++s) {
            int slot = tid + s * 256;
            if (slot < 640) {
                int kk = slot / 20, c4 = slot - kk * 20;  // 20 v4bf per 80-wide row
                rb[s] = *(const v4bf*)&Hsrc[(bofs + k0 + kk) * CIN + src_c + c4 * 4];
            }
        }
        if (k0 + 32 < N_)   // global_prefetch_b8 of the tile after next
            __builtin_prefetch(&A[(size_t)(m0 + (tid >> 1)) * N_ + k0 + 32], 0, 1);
    };
    auto sstore = [&](int buf) {
        __bf16* Ad = AsBuf + buf * (128 * LDK);
        __bf16* Bd = BtBuf + buf * (CAT * LDK);
#pragma unroll
        for (int s = 0; s < 4; ++s) {
            int slot = tid + s * 256;
            int r = slot >> 3, k4 = slot & 7;
            *(v4bf*)&Ad[r * LDK + k4 * 4] = cvt4(ra[s]);
        }
#pragma unroll
        for (int s = 0; s < 3; ++s) {
            int slot = tid + s * 256;
            if (slot < 640) {
                int kk = slot / 20, c4 = slot - kk * 20;
                Bd[(c4 * 4 + 0) * LDK + kk] = rb[s][0];   // transpose scatter
                Bd[(c4 * 4 + 1) * LDK + kk] = rb[s][1];
                Bd[(c4 * 4 + 2) * LDK + kk] = rb[s][2];
                Bd[(c4 * 4 + 3) * LDK + kk] = rb[s][3];
            }
        }
    };

    v8f acc[5] = {};
    gload(0);
    for (int k0 = 0; k0 < N_; k0 += 32) {
        const int buf = (k0 >> 5) & 1;
        sstore(buf);
        __syncthreads();                 // one barrier/iter: orders compute(i) vs sstore(i+2)
        if (k0 + 32 < N_) gload(k0 + 32);
        const __bf16* Ab = AsBuf + buf * (128 * LDK);
        const __bf16* Bb = BtBuf + buf * (CAT * LDK);
        // preload ALL fragments and pin them, then fire the WMMAs back-to-back
        v16bf af = frag_ld(&Ab[(wv * 16 + (ln & 15)) * LDK], koff);
        v16bf bfr[5];
#pragma unroll
        for (int t = 0; t < 5; ++t)
            bfr[t] = frag_ld(&Bb[(t * 16 + (ln & 15)) * LDK], koff);
        pin(af);
#pragma unroll
        for (int t = 0; t < 5; ++t) pin(bfr[t]);
#pragma unroll
        for (int t = 0; t < 5; ++t)
            acc[t] = __builtin_amdgcn_wmma_f32_16x16x32_bf16(
                false, af, false, bfr[t], (short)0, acc[t], false, false);
    }

    // Epilogue: stage 128x80 bf16 output tile in LDS, store coalesced b64.
    __syncthreads();                     // all compute reads of smem done
    __bf16* Ot = smem;                   // [128][88]; 88*2=176B row stride (8B aligned)
    const int nn = ln & 15;
#pragma unroll
    for (int t = 0; t < 5; ++t)
#pragma unroll
        for (int r = 0; r < 8; ++r)      // D: VGPR r -> M=r (lanes<16) / 8+r; N=lane&15
            Ot[(wv * 16 + 8 * hl + r) * 88 + t * 16 + nn] = (__bf16)acc[t][r];
    __syncthreads();
#pragma unroll
    for (int s = 0; s < 10; ++s) {       // 128 rows x 20 b64-chunks = 2560 slots
        int slot = tid + s * 256;
        int r = slot / 20, c4 = slot - r * 20;
        *(unsigned long long*)&Hdst[(bofs + m0 + r) * CIN + dst_c + c4 * 4] =
            *(const unsigned long long*)&Ot[r * 88 + c4 * 4];
    }
}

// ---------------------------------------------------------------------------
// linear_gemm: Y = act(H[BN,720] @ W[720, NT*16] + bias), fused GRU epilogue
//   MODE 0: ZR = sigmoid(.)               (gate, NT=8 -> 128 cols)
//   MODE 1: out = r*state + (1-r)*tanh(.) (update, NT=4 -> 64 cols)
// ---------------------------------------------------------------------------
template<int NT, int MODE>
__global__ __launch_bounds__(256) void linear_gemm(
    const __bf16* __restrict__ H, const float* __restrict__ W,
    const float* __restrict__ bias, float* __restrict__ ZR,
    const float* __restrict__ state, float* __restrict__ out) {
    constexpr int NC = NT * 16;
    constexpr int WSLOT = NC * 8;        // 32 x NC f32, as float4 slots
    constexpr int SW = WSLOT / 256;      // 4 (gate) / 2 (update)
    __shared__ __align__(16) __bf16 smem[2 * 128 * LDK + 2 * NC * LDK];
    __bf16* AsBuf = smem;
    __bf16* WtBuf = smem + 2 * 128 * LDK;

    const int tid  = threadIdx.x;
    const int wv   = tid >> 5;
    const int ln   = tid & 31;
    const int hl   = ln >> 4;
    const int koff = hl * 8;
    const int r0   = blockIdx.x * 128;

    v4bf rah[4];   // staged H rows: 128x32 bf16 = 1024 v4bf slots
    v4bf rw[SW];   // staged W (f32->bf16)

    auto gload = [&](int k0) {
#pragma unroll
        for (int s = 0; s < 4; ++s) {
            int slot = tid + s * 256;
            int r = slot >> 3, k4 = slot & 7;
            int k = k0 + k4 * 4;                     // 720 % 4 == 0: chunk all-or-nothing
            rah[s] = (k < CIN) ? *(const v4bf*)&H[(size_t)(r0 + r) * CIN + k] : zero4();
        }
#pragma unroll
        for (int s = 0; s < SW; ++s) {
            int slot = tid + s * 256;
            int kk = slot / (NC / 4), c4 = slot - kk * (NC / 4);
            int k = k0 + kk;
            rw[s] = (k < CIN) ? cvt4(*(const float4*)&W[(size_t)k * NC + c4 * 4]) : zero4();
        }
    };
    auto sstore = [&](int buf) {
        __bf16* Ad = AsBuf + buf * (128 * LDK);
        __bf16* Wd = WtBuf + buf * (NC * LDK);
#pragma unroll
        for (int s = 0; s < 4; ++s) {
            int slot = tid + s * 256;
            int r = slot >> 3, k4 = slot & 7;
            *(v4bf*)&Ad[r * LDK + k4 * 4] = rah[s];
        }
#pragma unroll
        for (int s = 0; s < SW; ++s) {
            int slot = tid + s * 256;
            int kk = slot / (NC / 4), c4 = slot - kk * (NC / 4);
            Wd[(c4 * 4 + 0) * LDK + kk] = rw[s][0];
            Wd[(c4 * 4 + 1) * LDK + kk] = rw[s][1];
            Wd[(c4 * 4 + 2) * LDK + kk] = rw[s][2];
            Wd[(c4 * 4 + 3) * LDK + kk] = rw[s][3];
        }
    };

    v8f acc[NT] = {};
    gload(0);
    for (int k0 = 0; k0 < 736; k0 += 32) {           // ceil(720/32)*32
        const int buf = (k0 >> 5) & 1;
        sstore(buf);
        __syncthreads();
        if (k0 + 32 < 736) gload(k0 + 32);
        const __bf16* Ab = AsBuf + buf * (128 * LDK);
        const __bf16* Wb = WtBuf + buf * (NC * LDK);
        v16bf af = frag_ld(&Ab[(wv * 16 + (ln & 15)) * LDK], koff);
        v16bf bfr[NT];
#pragma unroll
        for (int t = 0; t < NT; ++t)
            bfr[t] = frag_ld(&Wb[(t * 16 + (ln & 15)) * LDK], koff);
        pin(af);
#pragma unroll
        for (int t = 0; t < NT; ++t) pin(bfr[t]);
#pragma unroll
        for (int t = 0; t < NT; ++t)
            acc[t] = __builtin_amdgcn_wmma_f32_16x16x32_bf16(
                false, af, false, bfr[t], (short)0, acc[t], false, false);
    }

    const int nn = ln & 15;
#pragma unroll
    for (int t = 0; t < NT; ++t)
#pragma unroll
        for (int r = 0; r < 8; ++r) {
            int row = r0 + wv * 16 + 8 * hl + r;
            int col = t * 16 + nn;
            float v = acc[t][r] + bias[col];
            if (MODE == 0) {
                ZR[(size_t)row * 128 + col] = 1.0f / (1.0f + __expf(-v));
            } else {
                float hc = tanhf(v);
                float rr = ZR[(size_t)row * 128 + 64 + col];   // r = ZR[:, 64:]
                float st = state[(size_t)row * DOUT + col];
                out[(size_t)row * DOUT + col] = rr * st + (1.0f - rr) * hc;
            }
        }
}

// ---------------------------------------------------------------------------
extern "C" void kernel_launch(void* const* d_in, const int* in_sizes, int n_in,
                              void* d_out, int out_size, void* d_ws, size_t ws_size,
                              hipStream_t stream) {
    const float* x     = (const float*)d_in[0];
    const float* state = (const float*)d_in[1];
    const float* sup   = (const float*)d_in[2];
    const float* gW    = (const float*)d_in[3];
    const float* gb    = (const float*)d_in[4];
    const float* uW    = (const float*)d_in[5];
    const float* ub    = (const float*)d_in[6];
    float* out = (float*)d_out;

    // workspace: H (bf16 concat buffer, 94.4MB) | ZR (f32, 33.6MB)
    __bf16* H  = (__bf16*)d_ws;
    float*  ZR = (float*)((char*)d_ws + (size_t)BN * CIN * sizeof(__bf16));

    dim3 packGrid((BN * CAT + 255) / 256);
    dim3 propGrid(N_ / 128, B_);
    dim3 linGrid(BN / 128);

    // ---- gate GCN ----
    pack_kernel<false><<<packGrid, 256, 0, stream>>>(x, state, nullptr, H);
    for (int s = 0; s < S_; ++s) {
        const float* A = sup + (size_t)s * N_ * N_;
        prop_gemm<<<propGrid, 256, 0, stream>>>(A, H, H, 0,                 (2 * s + 1) * CAT);
        prop_gemm<<<propGrid, 256, 0, stream>>>(A, H, H, (2 * s + 1) * CAT, (2 * s + 2) * CAT);
    }
    linear_gemm<8, 0><<<linGrid, 256, 0, stream>>>(H, gW, gb, ZR, nullptr, nullptr);

    // ---- candidate GCN ----
    pack_kernel<true><<<packGrid, 256, 0, stream>>>(x, state, ZR, H);
    for (int s = 0; s < S_; ++s) {
        const float* A = sup + (size_t)s * N_ * N_;
        prop_gemm<<<propGrid, 256, 0, stream>>>(A, H, H, 0,                 (2 * s + 1) * CAT);
        prop_gemm<<<propGrid, 256, 0, stream>>>(A, H, H, (2 * s + 1) * CAT, (2 * s + 2) * CAT);
    }
    linear_gemm<4, 1><<<linGrid, 256, 0, stream>>>(H, uW, ub, ZR, state, out);
}
